// MultiheadAttention_67946382622992
// MI455X (gfx1250) — matrix operand order repaired and checked by reading, not compiled, not run on hardware
//
#include <hip/hip_runtime.h>

#define EMBED 1024
#define NHEAD 16
#define HDIM  64
#define BB    4
#define SS    2048

typedef __attribute__((ext_vector_type(16))) __bf16 v16bf;
typedef __attribute__((ext_vector_type(8)))  float  v8f;
typedef __attribute__((ext_vector_type(4)))  unsigned int v4u;
typedef __attribute__((ext_vector_type(8)))  int    v8i;
typedef __attribute__((ext_vector_type(4)))  int    v4i;

union FragBF {
  v16bf  v;
  __bf16 h[16];
  uint4  u[2];
};
union Pack8 { __bf16 h[8]; uint4 u; };
union Pack4 { __bf16 h[4]; uint2 u; };

static __device__ __forceinline__ v8f zero8() {
  v8f z;
#pragma unroll
  for (int i = 0; i < 8; ++i) z[i] = 0.0f;
  return z;
}

static __device__ __forceinline__ v8f wmma_bf16(const FragBF& a, const FragBF& b, v8f c) {
  // D = A(16x32 bf16) x B(32x16 bf16) + C(16x16 f32)
  return __builtin_amdgcn_wmma_f32_16x16x32_bf16(false, a.v, false, b.v, (short)0, c,
                                                 false, false);
}

static __device__ __forceinline__ float rowMax16(float v) {
  v = fmaxf(v, __shfl_xor(v, 1, 32));
  v = fmaxf(v, __shfl_xor(v, 2, 32));
  v = fmaxf(v, __shfl_xor(v, 4, 32));
  v = fmaxf(v, __shfl_xor(v, 8, 32));
  return v;
}

// ---- Tensor Data Mover: 2-D tile (bf16 elements) global -> LDS ----
// tile_d0 elements contiguous (dim0), tile_d1 rows with stride0 elements.
static __device__ __forceinline__ void tdm_load_2d(unsigned int lds_addr,
                                                   const __bf16* gptr,
                                                   unsigned int tensor_d0,
                                                   unsigned int tensor_d1,
                                                   unsigned int tile_d0,
                                                   unsigned int tile_d1,
                                                   unsigned int stride0) {
  unsigned long long ga = (unsigned long long)(uintptr_t)gptr;
  v4u g0;
  g0[0] = 1u;                                                // count=1, user mode
  g0[1] = lds_addr;                                          // LDS byte address
  g0[2] = (unsigned int)ga;                                  // global_addr[31:0]
  g0[3] = (unsigned int)((ga >> 32) & 0x1FFFFFFu) | (2u << 30);  // addr[56:32] | type=2
  v8i g1;
  g1[0] = (int)(1u << 16);                                   // data_size=1 (2 bytes)
  g1[1] = (int)((tensor_d0 & 0xFFFFu) << 16);                // tensor_dim0[15:0]
  g1[2] = (int)(((tensor_d0 >> 16) & 0xFFFFu) |              // tensor_dim0[31:16]
                ((tensor_d1 & 0xFFFFu) << 16));              // tensor_dim1[15:0]
  g1[3] = (int)(((tensor_d1 >> 16) & 0xFFFFu) |              // tensor_dim1[31:16]
                ((tile_d0 & 0xFFFFu) << 16));                // tile_dim0
  g1[4] = (int)(tile_d1 & 0xFFFFu);                          // tile_dim1, tile_dim2=0
  g1[5] = (int)stride0;                                      // tensor_dim0_stride[31:0]
  g1[6] = 0;                                                 // stride hi, dim1_stride lo
  g1[7] = 0;
  v4i z4 = {0, 0, 0, 0};
#if __clang_major__ >= 23
  v8i z8 = {0, 0, 0, 0, 0, 0, 0, 0};
  __builtin_amdgcn_tensor_load_to_lds(g0, g1, z4, z4, z8, 0);
#else
  __builtin_amdgcn_tensor_load_to_lds(g0, g1, z4, z4, 0);
#endif
}

static __device__ __forceinline__ unsigned int lds_off(const void* p) {
  return (unsigned int)(uintptr_t)p;  // LDS aperture keeps the offset in the low 32 bits
}

// ---------------- weight conversion fp32 -> bf16 (x4 vectorized) ----------------
__global__ __launch_bounds__(256) void mha_cvt_bf16x4_kernel(
    const float* __restrict__ src, __bf16* __restrict__ dst, int n4) {
  int i = blockIdx.x * 256 + threadIdx.x;
  if (i >= n4) return;
  float4 f = ((const float4*)src)[i];
  Pack4 pk;
  pk.h[0] = (__bf16)f.x; pk.h[1] = (__bf16)f.y;
  pk.h[2] = (__bf16)f.z; pk.h[3] = (__bf16)f.w;
  ((uint2*)dst)[i] = pk.u;
}

// ---------------- WMMA GEMM: Y[M=8192,N=1024] = X @ W^T + bias ----------------
// Block: 8 waves, each wave 32 rows x 64 cols; W tile staged in LDS by the TDM.
// MODE 0: X=fp32, out bf16 [b,h,s,d], scaled (Q)
// MODE 1: X=fp32, out bf16 [b,h,s,d]           (K)
// MODE 2: X=fp32, out bf16 [b,h,d,s] transposed (V)
// MODE 3: X=bf16, out fp32 row-major            (O projection -> d_out)
template <int MODE>
__global__ __launch_bounds__(256) void mha_gemm_kernel(
    const float* __restrict__ Xf, const __bf16* __restrict__ Xh,
    const __bf16* __restrict__ W, const float* __restrict__ bias,
    __bf16* __restrict__ outH, float* __restrict__ outF, float scale) {
  __shared__ __align__(16) __bf16 Wlds[2][64 * 32];  // 64 weight rows x 32 k, x2 buffers

  const int lane = threadIdx.x & 31;
  const int wave = threadIdx.x >> 5;
  const int n16  = lane & 15;
  const int half = lane >> 4;
  const int mRow0   = (blockIdx.x * 8 + wave) * 32;  // 32 rows per wave
  const int colBase = blockIdx.y * 64;               // 64 cols per block

  v8f acc[8];
#pragma unroll
  for (int t = 0; t < 8; ++t) acc[t] = zero8();

  const int arow0 = mRow0 + n16;
  const int arow1 = mRow0 + 16 + n16;
  const __bf16* wTile = W + (size_t)colBase * EMBED;  // rows colBase..colBase+63

  const int NSTEP = EMBED / 32;
  if (wave == 0) {
    tdm_load_2d(lds_off(&Wlds[0][0]), wTile, EMBED, EMBED, 32, 64, EMBED);
  }

  for (int s = 0; s < NSTEP; ++s) {
    const int k0 = s * 32;
    if (wave == 0) {
      if (s + 1 < NSTEP) {
        tdm_load_2d(lds_off(&Wlds[(s + 1) & 1][0]), wTile + (s + 1) * 32,
                    EMBED, EMBED, 32, 64, EMBED);
        __builtin_amdgcn_s_wait_tensorcnt(1);  // tile s finished, tile s+1 in flight
      } else {
        __builtin_amdgcn_s_wait_tensorcnt(0);  // last tile finished
      }
    }
    __syncthreads();  // W tile s is resident in Wlds[s&1]

    FragBF a0, a1;
    if (MODE == 3) {
      const __bf16* p0 = Xh + (size_t)arow0 * EMBED + k0 + 8 * half;
      a0.u[0] = *(const uint4*)(p0);
      a0.u[1] = *(const uint4*)(p0 + 16);
      const __bf16* p1 = Xh + (size_t)arow1 * EMBED + k0 + 8 * half;
      a1.u[0] = *(const uint4*)(p1);
      a1.u[1] = *(const uint4*)(p1 + 16);
    } else {
#pragma unroll
      for (int g = 0; g < 2; ++g) {
        const float* p = Xf + (size_t)(g ? arow1 : arow0) * EMBED + k0 + 8 * half;
        float4 f0 = ((const float4*)p)[0];
        float4 f1 = ((const float4*)p)[1];
        float4 f2 = ((const float4*)(p + 16))[0];
        float4 f3 = ((const float4*)(p + 16))[1];
        FragBF& a = g ? a1 : a0;
        a.h[0]  = (__bf16)f0.x; a.h[1]  = (__bf16)f0.y; a.h[2]  = (__bf16)f0.z; a.h[3]  = (__bf16)f0.w;
        a.h[4]  = (__bf16)f1.x; a.h[5]  = (__bf16)f1.y; a.h[6]  = (__bf16)f1.z; a.h[7]  = (__bf16)f1.w;
        a.h[8]  = (__bf16)f2.x; a.h[9]  = (__bf16)f2.y; a.h[10] = (__bf16)f2.z; a.h[11] = (__bf16)f2.w;
        a.h[12] = (__bf16)f3.x; a.h[13] = (__bf16)f3.y; a.h[14] = (__bf16)f3.z; a.h[15] = (__bf16)f3.w;
      }
    }

    const __bf16* wb = &Wlds[s & 1][0];
#pragma unroll
    for (int t = 0; t < 4; ++t) {
      FragBF b;  // B[k][n] = W[colBase + t*16 + n][k0 + k]
      const __bf16* p = wb + (t * 16 + n16) * 32 + 8 * half;
      b.u[0] = ((const uint4*)p)[0];
      b.u[1] = ((const uint4*)p)[2];
      acc[t]     = wmma_bf16(a0, b, acc[t]);
      acc[4 + t] = wmma_bf16(a1, b, acc[4 + t]);
    }
    __syncthreads();  // reads done before buffer is re-filled
  }

#pragma unroll
  for (int g = 0; g < 2; ++g) {
    const int mBase = mRow0 + 16 * g;
#pragma unroll
    for (int t = 0; t < 4; ++t) {
      const v8f& a = acc[4 * g + t];
      const int col = colBase + t * 16 + n16;
      const float bc = bias[col];
      if (MODE == 0 || MODE == 1) {
        const int h = col >> 6, d = col & 63;
#pragma unroll
        for (int r = 0; r < 8; ++r) {
          const int m = mBase + r + 8 * half;
          const int b = m >> 11, s = m & 2047;
          float y = (a[r] + bc) * scale;
          outH[(((size_t)b * NHEAD + h) * SS + s) * HDIM + d] = (__bf16)y;
        }
      } else if (MODE == 2) {
        const int h = col >> 6, d = col & 63;
        const int m0 = mBase + 8 * half;
        const int b = m0 >> 11, s0 = m0 & 2047;
        Pack8 pk;
#pragma unroll
        for (int r = 0; r < 8; ++r) pk.h[r] = (__bf16)(a[r] + bc);
        *(uint4*)&outH[(((size_t)b * NHEAD + h) * HDIM + d) * SS + s0] = pk.u;
      } else {
#pragma unroll
        for (int r = 0; r < 8; ++r) {
          const int m = mBase + r + 8 * half;
          outF[(size_t)m * EMBED + col] = a[r] + bc;
        }
      }
    }
  }
}

// ---------------- flash attention: one wave per (head, 16-query block) ----------------
// Qh/Kh: bf16 [b,h,s,64] (Q pre-scaled by 1/8); Vt: bf16 [b,h,64,s]; Ao: bf16 [b,s,h*64+d]
// K/V tiles streamed by the TDM (double buffered); row sums via wmma(P, ones).
__global__ __launch_bounds__(32) void mha_flash_kernel(
    const __bf16* __restrict__ Qh, const __bf16* __restrict__ Kh,
    const __bf16* __restrict__ Vt, __bf16* __restrict__ Ao) {
  __shared__ __align__(16) __bf16 Kls[2][32 * 64];  // [key][d]
  __shared__ __align__(16) __bf16 Vls[2][64 * 32];  // [d][key]
  __shared__ __align__(16) __bf16 Pl[16 * 32];

  const int lane  = threadIdx.x & 31;
  const int n16   = lane & 15;
  const int half  = lane >> 4;
  const int qBase = blockIdx.x * 16;
  const int bh    = blockIdx.y;

  const size_t hb = (size_t)bh * SS * HDIM;
  const __bf16* Qp = Qh + hb;
  const __bf16* Kp = Kh + hb;
  const __bf16* Vp = Vt + hb;

  // Q A-fragments over D: q0 covers d 0..31, q1 covers d 32..63
  FragBF q0, q1;
  {
    const __bf16* p = Qp + (size_t)(qBase + n16) * HDIM + 16 * half;
    q0.u[0] = *(const uint4*)(p);
    q0.u[1] = *(const uint4*)(p + 8);
    q1.u[0] = *(const uint4*)(p + 32);
    q1.u[1] = *(const uint4*)(p + 40);
  }
  FragBF ones;
#pragma unroll
  for (int i = 0; i < 16; ++i) ones.h[i] = (__bf16)1.0f;

  v8f O[4];
#pragma unroll
  for (int t = 0; t < 4; ++t) O[t] = zero8();
  float M[8], L[8];
#pragma unroll
  for (int r = 0; r < 8; ++r) { M[r] = -3.0e38f; L[r] = 0.0f; }

  const int nblk = (qBase + 16 + 31) / 32;  // number of 32-key blocks (causal)

  // prologue: stream first K/V tiles
  tdm_load_2d(lds_off(&Kls[0][0]), Kp, HDIM, SS, HDIM, 32, HDIM);  // 32 keys x 64 d
  tdm_load_2d(lds_off(&Vls[0][0]), Vp, SS, HDIM, 32, HDIM, SS);    // 64 d x 32 keys

  for (int s = 0; s < nblk; ++s) {
    const int j0 = s * 32;
    if (s + 1 < nblk) {
      const int j1 = (s + 1) * 32;
      tdm_load_2d(lds_off(&Kls[(s + 1) & 1][0]), Kp + (size_t)j1 * HDIM,
                  HDIM, SS, HDIM, 32, HDIM);
      tdm_load_2d(lds_off(&Vls[(s + 1) & 1][0]), Vp + j1,
                  SS, HDIM, 32, HDIM, SS);
      __builtin_amdgcn_s_wait_tensorcnt(2);  // current pair done, next pair in flight
    } else {
      __builtin_amdgcn_s_wait_tensorcnt(0);  // last pair done
    }
    const __bf16* kb_base = &Kls[s & 1][0];
    const __bf16* vb_base = &Vls[s & 1][0];

    // ---- scores: two 16x16 tiles (keys j0..j0+15, j0+16..j0+31) ----
    v8f c0 = zero8(), c1 = zero8();
    {
      FragBF kb;
      const __bf16* p = kb_base + n16 * HDIM + 16 * half;  // keys 0..15 of the tile
      kb.u[0] = ((const uint4*)p)[0];        kb.u[1] = ((const uint4*)p)[1];
      c0 = wmma_bf16(q0, kb, c0);
      kb.u[0] = ((const uint4*)(p + 32))[0]; kb.u[1] = ((const uint4*)(p + 32))[1];
      c0 = wmma_bf16(q1, kb, c0);
      const __bf16* p2 = p + 16 * HDIM;                    // keys 16..31
      kb.u[0] = ((const uint4*)p2)[0];        kb.u[1] = ((const uint4*)p2)[1];
      c1 = wmma_bf16(q0, kb, c1);
      kb.u[0] = ((const uint4*)(p2 + 32))[0]; kb.u[1] = ((const uint4*)(p2 + 32))[1];
      c1 = wmma_bf16(q1, kb, c1);
    }

    // ---- causal mask (only near the diagonal) ----
    if (j0 + 31 > qBase) {
#pragma unroll
      for (int r = 0; r < 8; ++r) {
        const int qrow = qBase + r + 8 * half;
        if (j0 + n16 > qrow)      c0[r] = -1.0e30f;
        if (j0 + 16 + n16 > qrow) c1[r] = -1.0e30f;
      }
    }

    // ---- online softmax: max via lane shuffles, P staged to LDS ----
    float alpha[8];
#pragma unroll
    for (int r = 0; r < 8; ++r) {
      float vmax = rowMax16(fmaxf(c0[r], c1[r]));
      float Mn = fmaxf(M[r], vmax);
      alpha[r] = __expf(M[r] - Mn);
      M[r] = Mn;
      const int row = r + 8 * half;
      Pl[row * 32 + n16]      = (__bf16)__expf(c0[r] - Mn);
      Pl[row * 32 + 16 + n16] = (__bf16)__expf(c1[r] - Mn);
    }
#pragma unroll
    for (int t = 0; t < 4; ++t) {
#pragma unroll
      for (int r = 0; r < 8; ++r) O[t][r] = O[t][r] * alpha[r];
    }
    __syncthreads();  // single wave: LDS ordering via dscnt

    // ---- re-layout P (C-format in LDS) as an A-fragment 16x32 ----
    FragBF pa;
    {
      const __bf16* ps = &Pl[n16 * 32 + 16 * half];
      pa.u[0] = ((const uint4*)ps)[0];
      pa.u[1] = ((const uint4*)ps)[1];  // +8 elements
    }
    // ---- row sums on the matrix pipe: ls = P @ ones (each column = rowsum) ----
    v8f ls = wmma_bf16(pa, ones, zero8());
#pragma unroll
    for (int r = 0; r < 8; ++r) L[r] = L[r] * alpha[r] + ls[r];

    // ---- O += P @ V  (B[k][n] = Vls[d=tileN][key], contiguous over keys) ----
#pragma unroll
    for (int t = 0; t < 4; ++t) {
      FragBF vb;
      const __bf16* pv = vb_base + (t * 16 + n16) * 32 + 16 * half;
      vb.u[0] = ((const uint4*)pv)[0];
      vb.u[1] = ((const uint4*)pv)[1];
      O[t] = wmma_bf16(pa, vb, O[t]);
    }
    __syncthreads();
  }

  // ---- normalize and store concat-head layout [b, s, h*64 + d] ----
  const int b = bh >> 4, h = bh & 15;
#pragma unroll
  for (int r = 0; r < 8; ++r) {
    const float inv = 1.0f / L[r];
    const int s = qBase + r + 8 * half;
#pragma unroll
    for (int t = 0; t < 4; ++t) {
      Ao[((size_t)b * SS + s) * EMBED + h * HDIM + t * 16 + n16] =
          (__bf16)(O[t][r] * inv);
    }
  }
}

// ---------------- host-side orchestration ----------------
extern "C" void kernel_launch(void* const* d_in, const int* in_sizes, int n_in,
                              void* d_out, int out_size, void* d_ws, size_t ws_size,
                              hipStream_t stream) {
  (void)in_sizes; (void)n_in; (void)out_size; (void)ws_size;
  const float* q   = (const float*)d_in[0];
  const float* k   = (const float*)d_in[1];
  const float* v   = (const float*)d_in[2];
  // d_in[3] = causal mask (int32) -- applied analytically in the kernel
  const float* q_w = (const float*)d_in[4];
  const float* q_b = (const float*)d_in[5];
  const float* k_w = (const float*)d_in[6];
  const float* k_b = (const float*)d_in[7];
  const float* v_w = (const float*)d_in[8];
  const float* v_b = (const float*)d_in[9];
  const float* o_w = (const float*)d_in[10];
  const float* o_b = (const float*)d_in[11];
  float* out = (float*)d_out;

  // workspace (bf16 elements): 4 weights (1M each) + Qh/Kh/Vt/Ao (8M each) = 72 MB
  __bf16* Wq = (__bf16*)d_ws;
  __bf16* Wk = Wq + (size_t)EMBED * EMBED;
  __bf16* Wv = Wk + (size_t)EMBED * EMBED;
  __bf16* Wo = Wv + (size_t)EMBED * EMBED;
  __bf16* Qh = Wo + (size_t)EMBED * EMBED;
  __bf16* Kh = Qh + (size_t)BB * SS * EMBED;
  __bf16* Vt = Kh + (size_t)BB * SS * EMBED;
  __bf16* Ao = Vt + (size_t)BB * SS * EMBED;

  const int n4 = (EMBED * EMBED) / 4;
  const dim3 cg((n4 + 255) / 256), cb(256);
  mha_cvt_bf16x4_kernel<<<cg, cb, 0, stream>>>(q_w, Wq, n4);
  mha_cvt_bf16x4_kernel<<<cg, cb, 0, stream>>>(k_w, Wk, n4);
  mha_cvt_bf16x4_kernel<<<cg, cb, 0, stream>>>(v_w, Wv, n4);
  mha_cvt_bf16x4_kernel<<<cg, cb, 0, stream>>>(o_w, Wo, n4);

  // M = B*S = 8192 rows (256/block), N = 1024 cols (64/block)
  const dim3 gg((BB * SS) / 256, EMBED / 64), gb(256);
  mha_gemm_kernel<0><<<gg, gb, 0, stream>>>(q, nullptr, Wq, q_b, Qh, nullptr, 0.125f);
  mha_gemm_kernel<1><<<gg, gb, 0, stream>>>(k, nullptr, Wk, k_b, Kh, nullptr, 1.0f);
  mha_gemm_kernel<2><<<gg, gb, 0, stream>>>(v, nullptr, Wv, v_b, Vt, nullptr, 1.0f);

  mha_flash_kernel<<<dim3(SS / 16, BB * NHEAD), 32, 0, stream>>>(Qh, Kh, Vt, Ao);

  mha_gemm_kernel<3><<<gg, gb, 0, stream>>>(nullptr, Ao, Wo, o_b, nullptr, out, 1.0f);
}
